// SelfAttention_34634616275373
// MI455X (gfx1250) — matrix operand order repaired and checked by reading, not compiled, not run on hardware
//
#include <hip/hip_runtime.h>

typedef _Float16 h4  __attribute__((ext_vector_type(4)));
typedef _Float16 h8  __attribute__((ext_vector_type(8)));
typedef _Float16 h16 __attribute__((ext_vector_type(16)));
typedef float    f8  __attribute__((ext_vector_type(8)));

// ---------------------------------------------------------------------------
// WMMA helpers (CDNA5 gfx1250, wave32)
// ---------------------------------------------------------------------------
static __device__ __forceinline__ f8 wmma16(h16 a, h16 b, f8 c) {
  // D = A(16x32 f16) * B(32x16 f16) + C(16x16 f32)
  return __builtin_amdgcn_wmma_f32_16x16x32_f16(false, a, false, b, (short)0, c,
                                                false, false);
}

// A fragment: lane = 16*hh + m holds row m, K chunks [8h..8h+8) and [16+8h..)
static __device__ __forceinline__ h16 frag_a(const _Float16* base, int ld, int m, int hh) {
  const _Float16* p = base + m * ld + hh * 8;
  h8 lo = *(const h8*)(p);
  h8 hi = *(const h8*)(p + 16);
  return __builtin_shufflevector(lo, hi, 0,1,2,3,4,5,6,7,8,9,10,11,12,13,14,15);
}
// B fragment: lane = 16*hh + n holds column n, K chunks [8h..8h+8) and [16+8h..)
static __device__ __forceinline__ h16 frag_b(const _Float16* base, int ld, int n, int hh) {
  const _Float16* p = base + n * ld + hh * 8;
  h8 lo = *(const h8*)(p);
  h8 hi = *(const h8*)(p + 16);
  return __builtin_shufflevector(lo, hi, 0,1,2,3,4,5,6,7,8,9,10,11,12,13,14,15);
}

// ---------------------------------------------------------------------------
// Kernel 1: qkv = x @ w_in^T + b_in  (f32 in, f16 out), double-buffered LDS
//   M=4096 (B*S), N=3072 (3D), K=1024
//   Q,K columns -> qk16[4096][2048]; V columns -> vt16[b][h][dh][S] (transposed)
// ---------------------------------------------------------------------------
__global__ __launch_bounds__(256, 2)
void qkv_gemm_kernel(const float* __restrict__ x, const float* __restrict__ w_in,
                     const float* __restrict__ b_in,
                     _Float16* __restrict__ qk16, _Float16* __restrict__ vt16) {
  __shared__ __align__(16) _Float16 sA[2][128 * 32];
  __shared__ __align__(16) _Float16 sB[2][128 * 32];
  const int tid = threadIdx.x, lane = tid & 31, wv = tid >> 5;
  const int hh = lane >> 4, ln = lane & 15;
  const int mBase = blockIdx.y * 128, nBase = blockIdx.x * 128;
  const int wm = (wv >> 2) * 64, wn = (wv & 3) * 32;

  const f8 fz = {0.f,0.f,0.f,0.f,0.f,0.f,0.f,0.f};
  f8 acc[4][2];
#pragma unroll
  for (int i = 0; i < 4; ++i)
#pragma unroll
    for (int j = 0; j < 2; ++j) acc[i][j] = fz;

  // Per-thread staging coordinates: 1024 chunks of 4 floats per 128x32 tile.
  int srow[4], sc4[4];
#pragma unroll
  for (int j = 0; j < 4; ++j) {
    int c = tid + j * 256;
    srow[j] = c >> 3;
    sc4[j]  = (c & 7) * 4;
  }

  float4 va[4], vb[4];
  // Preload tile 0
#pragma unroll
  for (int j = 0; j < 4; ++j) {
    va[j] = *(const float4*)(x    + (size_t)(mBase + srow[j]) * 1024 + sc4[j]);
    vb[j] = *(const float4*)(w_in + (size_t)(nBase + srow[j]) * 1024 + sc4[j]);
  }
#pragma unroll
  for (int j = 0; j < 4; ++j) {
    h4 ha = {(_Float16)va[j].x, (_Float16)va[j].y, (_Float16)va[j].z, (_Float16)va[j].w};
    h4 hb = {(_Float16)vb[j].x, (_Float16)vb[j].y, (_Float16)vb[j].z, (_Float16)vb[j].w};
    *(h4*)(sA[0] + srow[j] * 32 + sc4[j]) = ha;
    *(h4*)(sB[0] + srow[j] * 32 + sc4[j]) = hb;
  }
  __syncthreads();

  for (int i = 0; i < 32; ++i) {
    const int cur = i & 1;
    const int kb = ((i + 1) & 31) * 32;    // branchless: last iter re-loads tile 0
    // Phase 1: issue next-tile global loads (kept in flight across the WMMAs)
#pragma unroll
    for (int j = 0; j < 4; ++j) {
      va[j] = *(const float4*)(x    + (size_t)(mBase + srow[j]) * 1024 + kb + sc4[j]);
      vb[j] = *(const float4*)(w_in + (size_t)(nBase + srow[j]) * 1024 + kb + sc4[j]);
    }
    __builtin_amdgcn_sched_barrier(0);
    // Phase 2: compute from current LDS buffer
    h16 af[4], bf[2];
#pragma unroll
    for (int tm = 0; tm < 4; ++tm) af[tm] = frag_a(sA[cur] + (wm + tm * 16) * 32, 32, ln, hh);
#pragma unroll
    for (int tn = 0; tn < 2; ++tn) bf[tn] = frag_b(sB[cur] + (wn + tn * 16) * 32, 32, ln, hh);
#pragma unroll
    for (int tm = 0; tm < 4; ++tm)
#pragma unroll
      for (int tn = 0; tn < 2; ++tn) acc[tm][tn] = wmma16(af[tm], bf[tn], acc[tm][tn]);
    __builtin_amdgcn_sched_barrier(0);
    // Phase 3: convert + store next tile into the spare buffer
#pragma unroll
    for (int j = 0; j < 4; ++j) {
      h4 ha = {(_Float16)va[j].x, (_Float16)va[j].y, (_Float16)va[j].z, (_Float16)va[j].w};
      h4 hb = {(_Float16)vb[j].x, (_Float16)vb[j].y, (_Float16)vb[j].z, (_Float16)vb[j].w};
      *(h4*)(sA[cur ^ 1] + srow[j] * 32 + sc4[j]) = ha;
      *(h4*)(sB[cur ^ 1] + srow[j] * 32 + sc4[j]) = hb;
    }
    __syncthreads();
  }

  if (nBase < 2048) {
    // Q (cols 0..1023) and K (cols 1024..2047): block-uniform path
#pragma unroll
    for (int tn = 0; tn < 2; ++tn) {
      int colg = nBase + wn + tn * 16 + ln;
      float bias = b_in[colg];
#pragma unroll
      for (int tm = 0; tm < 4; ++tm) {
#pragma unroll
        for (int r = 0; r < 8; ++r) {
          int rowg = mBase + wm + tm * 16 + hh * 8 + r;
          qk16[(size_t)rowg * 2048 + colg] = (_Float16)(acc[tm][tn][r] + bias);
        }
      }
    }
  } else {
    // V columns, stored transposed per head: vt16[b][h][dh][S].
    // A lane's 8 accumulator rows are consecutive sequence positions ->
    // one packed b128 store per tile.
#pragma unroll
    for (int tn = 0; tn < 2; ++tn) {
      int dcol = nBase + wn + tn * 16 + ln - 2048;
      int head = dcol >> 6, dd = dcol & 63;
      float bias = b_in[nBase + wn + tn * 16 + ln];
#pragma unroll
      for (int tm = 0; tm < 4; ++tm) {
        int row0 = mBase + wm + tm * 16 + hh * 8;    // multiple of 8, no 2048-cross
        int bb = row0 >> 11, ss0 = row0 & 2047;
        h8 hv = {(_Float16)(acc[tm][tn][0] + bias), (_Float16)(acc[tm][tn][1] + bias),
                 (_Float16)(acc[tm][tn][2] + bias), (_Float16)(acc[tm][tn][3] + bias),
                 (_Float16)(acc[tm][tn][4] + bias), (_Float16)(acc[tm][tn][5] + bias),
                 (_Float16)(acc[tm][tn][6] + bias), (_Float16)(acc[tm][tn][7] + bias)};
        *(h8*)(vt16 + (((size_t)bb * 16 + head) * 64 + dd) * 2048 + ss0) = hv;
      }
    }
  }
}

// ---------------------------------------------------------------------------
// Kernel 2: causal flash attention, one WG per (128 q rows, b, h)
//   8 waves x 16 q rows; 32-key tiles; online softmax; double-buffered K/V;
//   O -> o16[4096][1024] f16
// ---------------------------------------------------------------------------
__global__ __launch_bounds__(256, 2)
void attn_kernel(const _Float16* __restrict__ qk16, const _Float16* __restrict__ vt16,
                 _Float16* __restrict__ o16) {
  __shared__ __align__(16) _Float16 sQ[128 * 64];      // 16 KB
  __shared__ __align__(16) _Float16 sK[2][32 * 64];    //  8 KB
  __shared__ __align__(16) _Float16 sV[2][64 * 32];    //  8 KB (V^T tile: [dh][k])
  __shared__ __align__(16) _Float16 sP[8 * 16 * 32];   //  8 KB per-wave P bounce
  const int tid = threadIdx.x, lane = tid & 31, wv = tid >> 5;
  const int hh = lane >> 4, ln = lane & 15;
  const int qTile = blockIdx.x, bh = blockIdx.y;
  const int b = bh >> 4, head = bh & 15;
  const int sBase = qTile * 128;

  const int rowK = tid >> 3, c8  = (tid & 7) * 8;      // K tile: 32x64 halves
  const int rowV = tid >> 2, c8v = (tid & 3) * 8;      // V tile: 64x32 halves
  const _Float16* gK = qk16 + (size_t)(b * 2048 + rowK) * 2048 + 1024 + head * 64 + c8;
  const _Float16* gV = vt16 + (((size_t)b * 16 + head) * 64 + rowV) * 2048 + c8v;

  // Stage Q tile (immutable for the whole block)
#pragma unroll
  for (int j = 0; j < 4; ++j) {
    int c = tid + j * 256;
    int row = c >> 3, cq = (c & 7) * 8;
    *(uint4*)(sQ + row * 64 + cq) =
        *(const uint4*)(qk16 + (size_t)(b * 2048 + sBase + row) * 2048 + head * 64 + cq);
  }
  // Stage K/V tile 0
  *(uint4*)(sK[0] + rowK * 64 + c8)  = *(const uint4*)(gK);
  *(uint4*)(sV[0] + rowV * 32 + c8v) = *(const uint4*)(gV);
  __syncthreads();

  // Q fragments are loop-invariant: hoist.
  const h16 aq0 = frag_a(sQ + (wv * 16) * 64,      64, ln, hh);
  const h16 aq1 = frag_a(sQ + (wv * 16) * 64 + 32, 64, ln, hh);

  const int qRow0 = sBase + wv * 16;
  float m_run[8], l_run[8];
#pragma unroll
  for (int r = 0; r < 8; ++r) { m_run[r] = -1e30f; l_run[r] = 0.f; }
  const f8 fz = {0.f,0.f,0.f,0.f,0.f,0.f,0.f,0.f};
  f8 oc[4];
#pragma unroll
  for (int dt = 0; dt < 4; ++dt) oc[dt] = fz;

  const int kSteps = (qTile + 1) * 4;
  for (int ks = 0; ks < kSteps; ++ks) {
    const int kt = ks * 32;
    const int cur = ks & 1;
    // Branchless prefetch: last iteration re-reads the current tile (discarded).
    const int ktn = (ks + 1 < kSteps) ? kt + 32 : kt;
    uint4 rKv = *(const uint4*)(gK + (size_t)ktn * 2048);
    uint4 rVv = *(const uint4*)(gV + ktn);
    __builtin_amdgcn_sched_barrier(0);
    if (kt <= qRow0 + 15) {            // wave-uniform: EXEC stays all-ones
      const _Float16* bK = sK[cur];
      const _Float16* bV = sV[cur];
      f8 s0 = fz, s1 = fz;
      h16 bk00 = frag_b(bK + 0,            64, ln, hh);
      h16 bk01 = frag_b(bK + 32,           64, ln, hh);
      h16 bk10 = frag_b(bK + 16 * 64,      64, ln, hh);
      h16 bk11 = frag_b(bK + 16 * 64 + 32, 64, ln, hh);
      s0 = wmma16(aq0, bk00, s0);
      s0 = wmma16(aq1, bk01, s0);
      s1 = wmma16(aq0, bk10, s1);
      s1 = wmma16(aq1, bk11, s1);

      const int k0 = kt + ln, k1 = kt + 16 + ln;
      float p0[8], p1[8], alpha[8];
#pragma unroll
      for (int r = 0; r < 8; ++r) {
        int qg = qRow0 + hh * 8 + r;
        float v0 = (k0 <= qg) ? s0[r] * 0.125f : -1e30f;
        float v1 = (k1 <= qg) ? s1[r] * 0.125f : -1e30f;
        float t = fmaxf(v0, v1);
        t = fmaxf(t, __shfl_xor(t, 1, 32));
        t = fmaxf(t, __shfl_xor(t, 2, 32));
        t = fmaxf(t, __shfl_xor(t, 4, 32));
        t = fmaxf(t, __shfl_xor(t, 8, 32));
        float mn = fmaxf(m_run[r], t);
        float a  = __expf(m_run[r] - mn);
        p0[r] = __expf(v0 - mn);
        p1[r] = __expf(v1 - mn);
        float rs = p0[r] + p1[r];
        rs += __shfl_xor(rs, 1, 32);
        rs += __shfl_xor(rs, 2, 32);
        rs += __shfl_xor(rs, 4, 32);
        rs += __shfl_xor(rs, 8, 32);
        l_run[r] = l_run[r] * a + rs;
        m_run[r] = mn;
        alpha[r] = a;
      }
#pragma unroll
      for (int dt = 0; dt < 4; ++dt)
#pragma unroll
        for (int r = 0; r < 8; ++r) oc[dt][r] *= alpha[r];

      // C-layout -> A-layout via per-wave LDS bounce (DS is in-order per wave)
      _Float16* pw = sP + wv * 512;
#pragma unroll
      for (int r = 0; r < 8; ++r) {
        pw[(hh * 8 + r) * 32 + ln]      = (_Float16)p0[r];
        pw[(hh * 8 + r) * 32 + 16 + ln] = (_Float16)p1[r];
      }
      asm volatile("s_wait_dscnt 0x0" ::: "memory");
      h16 ap = frag_a(pw, 32, ln, hh);
#pragma unroll
      for (int dt = 0; dt < 4; ++dt) {
        h16 bvf = frag_b(bV + (dt * 16) * 32, 32, ln, hh);
        oc[dt] = wmma16(ap, bvf, oc[dt]);
      }
    }
    __builtin_amdgcn_sched_barrier(0);
    *(uint4*)(sK[cur ^ 1] + rowK * 64 + c8)  = rKv;
    *(uint4*)(sV[cur ^ 1] + rowV * 32 + c8v) = rVv;
    __syncthreads();
  }

  float inv[8];
#pragma unroll
  for (int r = 0; r < 8; ++r) inv[r] = 1.f / l_run[r];
#pragma unroll
  for (int dt = 0; dt < 4; ++dt) {
    int colg = head * 64 + dt * 16 + ln;
#pragma unroll
    for (int r = 0; r < 8; ++r) {
      int rowg = b * 2048 + sBase + wv * 16 + hh * 8 + r;
      o16[(size_t)rowg * 1024 + colg] = (_Float16)(oc[dt][r] * inv[r]);
    }
  }
}

// ---------------------------------------------------------------------------
// Kernel 3: out = O @ w_out^T + b_out  (f16 A, f32 B, f32 out), double-buffered
//   M=4096, N=1024, K=1024
// ---------------------------------------------------------------------------
__global__ __launch_bounds__(256, 2)
void out_gemm_kernel(const _Float16* __restrict__ o16, const float* __restrict__ w_out,
                     const float* __restrict__ b_out, float* __restrict__ out) {
  __shared__ __align__(16) _Float16 sA[2][128 * 32];
  __shared__ __align__(16) _Float16 sB[2][128 * 32];
  const int tid = threadIdx.x, lane = tid & 31, wv = tid >> 5;
  const int hh = lane >> 4, ln = lane & 15;
  const int mBase = blockIdx.y * 128, nBase = blockIdx.x * 128;
  const int wm = (wv >> 2) * 64, wn = (wv & 3) * 32;

  const f8 fz = {0.f,0.f,0.f,0.f,0.f,0.f,0.f,0.f};
  f8 acc[4][2];
#pragma unroll
  for (int i = 0; i < 4; ++i)
#pragma unroll
    for (int j = 0; j < 2; ++j) acc[i][j] = fz;

  int arow[2], ac8[2], brow[4], bc4[4];
#pragma unroll
  for (int j = 0; j < 2; ++j) {
    int c = tid + j * 256;               // 512 chunks of 8 halves
    arow[j] = c >> 2;
    ac8[j]  = (c & 3) * 8;
  }
#pragma unroll
  for (int j = 0; j < 4; ++j) {
    int c = tid + j * 256;               // 1024 chunks of 4 floats
    brow[j] = c >> 3;
    bc4[j]  = (c & 7) * 4;
  }

  uint4 ra[2]; float4 rb[4];
#pragma unroll
  for (int j = 0; j < 2; ++j)
    ra[j] = *(const uint4*)(o16 + (size_t)(mBase + arow[j]) * 1024 + ac8[j]);
#pragma unroll
  for (int j = 0; j < 4; ++j)
    rb[j] = *(const float4*)(w_out + (size_t)(nBase + brow[j]) * 1024 + bc4[j]);
#pragma unroll
  for (int j = 0; j < 2; ++j) *(uint4*)(sA[0] + arow[j] * 32 + ac8[j]) = ra[j];
#pragma unroll
  for (int j = 0; j < 4; ++j) {
    h4 hb = {(_Float16)rb[j].x, (_Float16)rb[j].y, (_Float16)rb[j].z, (_Float16)rb[j].w};
    *(h4*)(sB[0] + brow[j] * 32 + bc4[j]) = hb;
  }
  __syncthreads();

  for (int i = 0; i < 32; ++i) {
    const int cur = i & 1;
    const int kb = ((i + 1) & 31) * 32;  // branchless: last iter re-loads tile 0
#pragma unroll
    for (int j = 0; j < 2; ++j)
      ra[j] = *(const uint4*)(o16 + (size_t)(mBase + arow[j]) * 1024 + kb + ac8[j]);
#pragma unroll
    for (int j = 0; j < 4; ++j)
      rb[j] = *(const float4*)(w_out + (size_t)(nBase + brow[j]) * 1024 + kb + bc4[j]);
    __builtin_amdgcn_sched_barrier(0);
    h16 af[4], bf[2];
#pragma unroll
    for (int tm = 0; tm < 4; ++tm) af[tm] = frag_a(sA[cur] + (wm + tm * 16) * 32, 32, ln, hh);
#pragma unroll
    for (int tn = 0; tn < 2; ++tn) bf[tn] = frag_b(sB[cur] + (wn + tn * 16) * 32, 32, ln, hh);
#pragma unroll
    for (int tm = 0; tm < 4; ++tm)
#pragma unroll
      for (int tn = 0; tn < 2; ++tn) acc[tm][tn] = wmma16(af[tm], bf[tn], acc[tm][tn]);
    __builtin_amdgcn_sched_barrier(0);
#pragma unroll
    for (int j = 0; j < 2; ++j) *(uint4*)(sA[cur ^ 1] + arow[j] * 32 + ac8[j]) = ra[j];
#pragma unroll
    for (int j = 0; j < 4; ++j) {
      h4 hb = {(_Float16)rb[j].x, (_Float16)rb[j].y, (_Float16)rb[j].z, (_Float16)rb[j].w};
      *(h4*)(sB[cur ^ 1] + brow[j] * 32 + bc4[j]) = hb;
    }
    __syncthreads();
  }

#pragma unroll
  for (int tn = 0; tn < 2; ++tn) {
    int colg = nBase + wn + tn * 16 + ln;
    float bias = b_out[colg];
#pragma unroll
    for (int tm = 0; tm < 4; ++tm) {
#pragma unroll
      for (int r = 0; r < 8; ++r) {
        int rowg = mBase + wm + tm * 16 + hh * 8 + r;
        out[(size_t)rowg * 1024 + colg] = acc[tm][tn][r] + bias;
      }
    }
  }
}

// ---------------------------------------------------------------------------
// Launch: x, w_in, b_in, w_out, b_out -> out [2,2048,1024] f32
// Workspace: qk16 16.8MB + vt16 8.4MB + o16 8.4MB = ~33.6MB
// ---------------------------------------------------------------------------
extern "C" void kernel_launch(void* const* d_in, const int* in_sizes, int n_in,
                              void* d_out, int out_size, void* d_ws, size_t ws_size,
                              hipStream_t stream) {
  (void)in_sizes; (void)n_in; (void)out_size; (void)ws_size;
  const float* x     = (const float*)d_in[0];
  const float* w_in  = (const float*)d_in[1];
  const float* b_in  = (const float*)d_in[2];
  const float* w_out = (const float*)d_in[3];
  const float* b_out = (const float*)d_in[4];
  float* out = (float*)d_out;

  _Float16* qk16 = (_Float16*)d_ws;                       // [4096][2048]
  _Float16* vt16 = qk16 + (size_t)4096 * 2048;            // [2][16][64][2048]
  _Float16* o16  = vt16 + (size_t)2 * 16 * 64 * 2048;     // [4096][1024]

  qkv_gemm_kernel<<<dim3(24, 32), 256, 0, stream>>>(x, w_in, b_in, qk16, vt16);
  attn_kernel    <<<dim3(16, 32), 256, 0, stream>>>(qk16, vt16, o16);
  out_gemm_kernel<<<dim3(8, 32),  256, 0, stream>>>(o16, w_out, b_out, out);
}
